// CrossEntropyLoss_89842125898441
// MI455X (gfx1250) — matrix operand order repaired and checked by reading, not compiled, not run on hardware
//
#include <hip/hip_runtime.h>

// Problem constants (from reference setup_inputs)
#define N_ROWS   4096      // 2 * 2048
#define HDIM     2048
#define VOCAB    32000
#define IGNORE_L (-100LL)

// Main-kernel tiling
#define MT       64        // rows per workgroup (2 M-waves * 32 rows)
#define NSLICE   10        // vocab slices (partial-LSE merge at the end)
#define SLICE_V  3200      // VOCAB / NSLICE
#define VBLK     128       // columns per workgroup per iteration (4 V-waves * 32)
#define NBLK     25        // SLICE_V / VBLK

typedef __attribute__((ext_vector_type(16))) __bf16 v16bf;
typedef __attribute__((ext_vector_type(8)))  __bf16 v8bf;
typedef __attribute__((ext_vector_type(8)))  float  v8f;

// ---------------------------------------------------------------------------
// f32 -> (bf16 hi, bf16 lo) split:  v ~= hi + lo  with |lo| <= ulp(hi)/2
// ---------------------------------------------------------------------------
__global__ void ce_cvt_split(const float* __restrict__ src,
                             __bf16* __restrict__ hi,
                             __bf16* __restrict__ lo, int n) {
  int stride = gridDim.x * blockDim.x;
  for (int i = blockIdx.x * blockDim.x + threadIdx.x; i < n; i += stride) {
    float v = src[i];
    __bf16 h = (__bf16)v;
    hi[i] = h;
    lo[i] = (__bf16)(v - (float)h);
  }
}

// ---------------------------------------------------------------------------
// A fragment as loaded here (16-bit A 16x32, wave32; contiguous k, same order as B):
//   lanes 0-15 : M = lane,    K = {0..15}
//   lanes 16-31: M = lane-16, K = {16..31}
// -> per lane: two contiguous 16B loads at (k0+koff) and (k0+koff+8)
// ---------------------------------------------------------------------------
__device__ __forceinline__ v16bf load_a_frag(const __bf16* __restrict__ p) {
  union { v16bf v; v8bf h[2]; } u;
  u.h[0] = *reinterpret_cast<const v8bf*>(p);
  u.h[1] = *reinterpret_cast<const v8bf*>(p + 8);
  return u.v;
}

// ---------------------------------------------------------------------------
// Main fused kernel: bf16x3 GEMM tile + online logsumexp over a vocab slice.
// grid.x = (N_ROWS/MT) * NSLICE, block = 256 threads (8 waves, wave32).
// Wave grid: mi in {0,1} (32 rows each), vi in {0..3} (32 cols each).
// Outputs per-slice partials pm/ps: lse = m + log(s) after merging slices.
// ---------------------------------------------------------------------------
__global__ void __launch_bounds__(256)
ce_main(const __bf16* __restrict__ xhi, const __bf16* __restrict__ xlo,
        const __bf16* __restrict__ whi, const __bf16* __restrict__ wlo,
        float* __restrict__ pm, float* __restrict__ ps) {
  __shared__ float lm[4][MT];
  __shared__ float lsd[4][MT];

  const int lane = threadIdx.x & 31;
  const int wv   = threadIdx.x >> 5;
  const int mi   = wv >> 2;          // 0..1
  const int vi   = wv & 3;           // 0..3
  const int lh   = lane & 15;        // position within half-wave
  const int lg   = lane >> 4;        // which half-wave (0/1)

  const int ntm   = N_ROWS / MT;     // 64
  const int tm    = blockIdx.x % ntm;
  const int slice = blockIdx.x / ntm;
  const int rowbase = tm * MT + mi * 32;

  // Per-lane running (max, sumexp). Slot = mt*8 + r maps to row
  // mt*16 + r + lg*8 of this wave's 32-row strip (C-matrix layout).
  float rm_[16], rs_[16];
#pragma unroll
  for (int i = 0; i < 16; ++i) { rm_[i] = -__builtin_inff(); rs_[i] = 0.f; }

  for (int blk = 0; blk < NBLK; ++blk) {
    const int vb = slice * SLICE_V + blk * VBLK + vi * 32;

    v8f acc[2][2] = {};   // 2 M-tiles x 2 V-tiles of 16x16 f32

    for (int k0 = 0; k0 < HDIM; k0 += 32) {
      v16bf ahi[2], alo[2], bhi[2], blo[2];
#pragma unroll
      for (int mt = 0; mt < 2; ++mt) {
        const size_t ra = (size_t)(rowbase + mt * 16 + lh) * HDIM + k0 + lg * 16;
        ahi[mt] = load_a_frag(xhi + ra);
        alo[mt] = load_a_frag(xlo + ra);
      }
#pragma unroll
      for (int vt = 0; vt < 2; ++vt) {
        // B (32x16): lane = column, K contiguous per half-wave -> 32B load
        const size_t rb = (size_t)(vb + vt * 16 + lh) * HDIM + k0 + lg * 16;
        bhi[vt] = *reinterpret_cast<const v16bf*>(whi + rb);
        blo[vt] = *reinterpret_cast<const v16bf*>(wlo + rb);
        __builtin_prefetch(whi + rb + 32, 0, 0);   // next K-step of this row
      }
#pragma unroll
      for (int mt = 0; mt < 2; ++mt)
#pragma unroll
        for (int vt = 0; vt < 2; ++vt) {
          acc[mt][vt] = __builtin_amdgcn_wmma_f32_16x16x32_bf16(
              false, ahi[mt], false, bhi[vt], (short)0, acc[mt][vt], false, false);
          acc[mt][vt] = __builtin_amdgcn_wmma_f32_16x16x32_bf16(
              false, ahi[mt], false, blo[vt], (short)0, acc[mt][vt], false, false);
          acc[mt][vt] = __builtin_amdgcn_wmma_f32_16x16x32_bf16(
              false, alo[mt], false, bhi[vt], (short)0, acc[mt][vt], false, false);
        }
    }

    // Online (max, sumexp) update: each lane sees 2 columns per row slot.
#pragma unroll
    for (int mt = 0; mt < 2; ++mt)
#pragma unroll
      for (int r = 0; r < 8; ++r) {
        const int slot = mt * 8 + r;
#pragma unroll
        for (int vt = 0; vt < 2; ++vt) {
          float z  = acc[mt][vt][r];
          float mo = rm_[slot];
          float mn = fmaxf(mo, z);
          rs_[slot] = rs_[slot] * __expf(mo - mn) + __expf(z - mn);
          rm_[slot] = mn;
        }
      }
  }

  // Reduce (m,s) across the 16 lanes of each half-wave (wave32 shuffles).
#pragma unroll
  for (int off = 1; off < 16; off <<= 1) {
#pragma unroll
    for (int slot = 0; slot < 16; ++slot) {
      float om = __shfl_xor(rm_[slot], off, 32);
      float os = __shfl_xor(rs_[slot], off, 32);
      float mn = fmaxf(rm_[slot], om);
      rs_[slot] = rs_[slot] * __expf(rm_[slot] - mn) + os * __expf(om - mn);
      rm_[slot] = mn;
    }
  }

  if (lh == 0) {   // lanes 0 and 16 hold the reduced values for their rows
#pragma unroll
    for (int slot = 0; slot < 16; ++slot) {
      int rl = (slot >> 3) * 16 + (slot & 7) + lg * 8;  // 0..31 in this wave
      lm[vi][mi * 32 + rl]  = rm_[slot];
      lsd[vi][mi * 32 + rl] = rs_[slot];
    }
  }
  __syncthreads();

  // Merge the 4 V-waves and emit per-slice partials.
  if (threadIdx.x < MT) {
    int row = threadIdx.x;
    float m = lm[0][row], s = lsd[0][row];
#pragma unroll
    for (int vv = 1; vv < 4; ++vv) {
      float m2 = lm[vv][row], s2 = lsd[vv][row];
      float mn = fmaxf(m, m2);
      s = s * __expf(m - mn) + s2 * __expf(m2 - mn);
      m = mn;
    }
    int grow = tm * MT + row;
    pm[(size_t)slice * N_ROWS + grow] = m;
    ps[(size_t)slice * N_ROWS + grow] = s;
  }
}

// ---------------------------------------------------------------------------
// Exact-f32 target logit: one wave per row (tiny: 16 MFLOP total).
// ---------------------------------------------------------------------------
__global__ void ce_tgt(const float* __restrict__ x, const float* __restrict__ w,
                       const long long* __restrict__ labels,
                       float* __restrict__ tl) {
  int wid  = (blockIdx.x * blockDim.x + threadIdx.x) >> 5;
  int lane = threadIdx.x & 31;
  if (wid >= N_ROWS) return;
  long long t = labels[wid];
  float sum = 0.f;
  if (t != IGNORE_L) {
    const float* xr = x + (size_t)wid * HDIM;
    const float* wr = w + (size_t)t * HDIM;
    for (int j = lane; j < HDIM; j += 32) sum = fmaf(xr[j], wr[j], sum);
  }
#pragma unroll
  for (int off = 16; off > 0; off >>= 1) sum += __shfl_xor(sum, off, 32);
  if (lane == 0) tl[wid] = sum;
}

// ---------------------------------------------------------------------------
// Finalize: merge slice partials (exact expf/logf), mean over valid labels.
// ---------------------------------------------------------------------------
__global__ void ce_final(const float* __restrict__ pm, const float* __restrict__ ps,
                         const float* __restrict__ tl,
                         const long long* __restrict__ labels,
                         float* __restrict__ out) {
  __shared__ float ssum[256];
  __shared__ int   scnt[256];
  float acc = 0.f; int cnt = 0;
  for (int n = threadIdx.x; n < N_ROWS; n += 256) {
    long long t = labels[n];
    if (t == IGNORE_L) continue;
    float m = pm[n], s = ps[n];
    for (int sl = 1; sl < NSLICE; ++sl) {
      float m2 = pm[(size_t)sl * N_ROWS + n];
      float s2 = ps[(size_t)sl * N_ROWS + n];
      float mn = fmaxf(m, m2);
      s = s * expf(m - mn) + s2 * expf(m2 - mn);
      m = mn;
    }
    acc += (m + logf(s)) - tl[n];
    cnt += 1;
  }
  ssum[threadIdx.x] = acc; scnt[threadIdx.x] = cnt;
  __syncthreads();
  for (int off = 128; off > 0; off >>= 1) {
    if (threadIdx.x < off) {
      ssum[threadIdx.x] += ssum[threadIdx.x + off];
      scnt[threadIdx.x] += scnt[threadIdx.x + off];
    }
    __syncthreads();
  }
  if (threadIdx.x == 0) out[0] = ssum[0] / (float)(scnt[0] > 0 ? scnt[0] : 1);
}

// ---------------------------------------------------------------------------
extern "C" void kernel_launch(void* const* d_in, const int* in_sizes, int n_in,
                              void* d_out, int out_size, void* d_ws, size_t ws_size,
                              hipStream_t stream) {
  const float*     x      = (const float*)d_in[0];      // [2,2048,2048] f32
  const float*     w      = (const float*)d_in[1];      // [32000,2048] f32
  const long long* labels = (const long long*)d_in[2];  // [2,2048] i64
  float*           out    = (float*)d_out;

  // Workspace carve-out (~296 MB), 256B-aligned slabs.
  char*  ws = (char*)d_ws;
  size_t o  = 0;
  auto carve = [&](size_t bytes) -> char* {
    char* p = ws + o;
    o += (bytes + 255) & ~(size_t)255;
    return p;
  };
  __bf16* whi = (__bf16*)carve((size_t)VOCAB * HDIM * sizeof(__bf16));
  __bf16* wlo = (__bf16*)carve((size_t)VOCAB * HDIM * sizeof(__bf16));
  __bf16* xhi = (__bf16*)carve((size_t)N_ROWS * HDIM * sizeof(__bf16));
  __bf16* xlo = (__bf16*)carve((size_t)N_ROWS * HDIM * sizeof(__bf16));
  float*  pm  = (float*)carve((size_t)NSLICE * N_ROWS * sizeof(float));
  float*  psv = (float*)carve((size_t)NSLICE * N_ROWS * sizeof(float));
  float*  tl  = (float*)carve((size_t)N_ROWS * sizeof(float));

  // 1) split-precision conversion (one-time)
  ce_cvt_split<<<4096, 256, 0, stream>>>(w, whi, wlo, VOCAB * HDIM);
  ce_cvt_split<<<1024, 256, 0, stream>>>(x, xhi, xlo, N_ROWS * HDIM);

  // 2) exact-f32 target logits (independent of the GEMM path)
  ce_tgt<<<(N_ROWS * 32) / 256, 256, 0, stream>>>(x, w, labels, tl);

  // 3) fused bf16x3 WMMA GEMM + online per-slice logsumexp
  ce_main<<<(N_ROWS / MT) * NSLICE, 256, 0, stream>>>(xhi, xlo, whi, wlo, pm, psv);

  // 4) merge slices, subtract target logit, masked mean
  ce_final<<<1, 256, 0, stream>>>(pm, psv, tl, labels, out);
}
